// GAT_EFA_27805618274511
// MI455X (gfx1250) — matrix operand order repaired and checked by reading, not compiled, not run on hardware
//
#include <hip/hip_runtime.h>

typedef __attribute__((ext_vector_type(16))) _Float16 v16h;
typedef __attribute__((ext_vector_type(8)))  _Float16 v8h;
typedef __attribute__((ext_vector_type(2)))  _Float16 h2;
typedef __attribute__((ext_vector_type(8)))  float    v8f;

__device__ __forceinline__ float lrelu(float x) { return x > 0.f ? x : 0.2f * x; }

// ---------------------------------------------------------------------------
// setup: reduce tiled-input weights, convert edge_emb to f16
// sc layout: [0:32]=s1 [32:64]=b1 [64:96]=s2 [96:128]=b2
// nr layout: [0]=nw1e [32]=nw1o [64]=nb1 [96]=nw2s [128]=nb2
//            [160]=nw3e [192]=nw3o [224]=nb3 [256]=nw4s [288]=nb4
// ---------------------------------------------------------------------------
__global__ void setup_kernel(
    const float* __restrict__ ew1w, const float* __restrict__ ew1b,
    const float* __restrict__ ew2w, const float* __restrict__ ew2b,
    const float* __restrict__ nw1w, const float* __restrict__ nw1b,
    const float* __restrict__ nw2w, const float* __restrict__ nw2b,
    const float* __restrict__ nw3w, const float* __restrict__ nw3b,
    const float* __restrict__ nw4w, const float* __restrict__ nw4b,
    const float* __restrict__ eembw,
    float* __restrict__ sc, float* __restrict__ nr, _Float16* __restrict__ emb16)
{
    const int tid = threadIdx.x;
    if (tid < 32) {
        float s = 0.f;
        for (int j = 0; j < 10; ++j) s += ew1w[tid * 10 + j];
        sc[tid] = s; sc[32 + tid] = ew1b[tid];
        s = 0.f;
        for (int j = 0; j < 10; ++j) s += ew2w[tid * 10 + j];
        sc[64 + tid] = s; sc[96 + tid] = ew2b[tid];

        float e = 0.f, o = 0.f;
        for (int t = 0; t < 10; ++t) { e += nw1w[tid * 20 + 2 * t]; o += nw1w[tid * 20 + 2 * t + 1]; }
        nr[tid] = e; nr[32 + tid] = o; nr[64 + tid] = nw1b[tid];
        s = 0.f;
        for (int j = 0; j < 10; ++j) s += nw2w[tid * 10 + j];
        nr[96 + tid] = s; nr[128 + tid] = nw2b[tid];
        e = 0.f; o = 0.f;
        for (int t = 0; t < 10; ++t) { e += nw3w[tid * 20 + 2 * t]; o += nw3w[tid * 20 + 2 * t + 1]; }
        nr[160 + tid] = e; nr[192 + tid] = o; nr[224 + tid] = nw3b[tid];
        s = 0.f;
        for (int j = 0; j < 10; ++j) s += nw4w[tid * 10 + j];
        nr[256 + tid] = s; nr[288 + tid] = nw4b[tid];
    }
    for (int idx = tid; idx < 64 * 64; idx += 256) emb16[idx] = (_Float16)eembw[idx];
}

// ---------------------------------------------------------------------------
// per-head setup: wf -> f16, fbias = wf.b + bf, af copy, and reduced
// attention vectors w1r/w2r (folding ah.w into wh1/wh2)
// ---------------------------------------------------------------------------
__global__ void conv_head_kernel(
    const float* __restrict__ wh1w, const float* __restrict__ wh1b,
    const float* __restrict__ wh2w, const float* __restrict__ wh2b,
    const float* __restrict__ ahw,
    const float* __restrict__ wfw, const float* __restrict__ wfb,
    const float* __restrict__ bf,
    const float* __restrict__ afw_in, const float* __restrict__ afb_in,
    int ind,
    _Float16* __restrict__ wf16h, float* __restrict__ fbiash,
    float* __restrict__ afwh, float* __restrict__ afbh,
    float* __restrict__ w1r, float* __restrict__ w2r, float* __restrict__ c12)
{
    const int tid = threadIdx.x;
    for (int idx = tid; idx < 64 * 64; idx += 256) wf16h[idx] = (_Float16)wfw[idx];
    if (tid < 64) { fbiash[tid] = wfb[tid] + bf[tid]; afwh[tid] = afw_in[tid]; }
    if (tid == 0) {
        afbh[0] = afb_in[0];
        float s1 = 0.f, s2 = 0.f;
        for (int o = 0; o < 64; ++o) { s1 += ahw[o] * wh1b[o]; s2 += ahw[o] * wh2b[o]; }
        c12[0] = s1; c12[1] = s2;
    }
    for (int i = tid; i < ind; i += 256) {
        float s1 = 0.f, s2 = 0.f;
        for (int o = 0; o < 64; ++o) {
            s1 += ahw[o] * wh1w[o * ind + i];
            s2 += ahw[o] * wh2w[o * ind + i];
        }
        w1r[i] = s1; w2r[i] = s2;
    }
}

// ---------------------------------------------------------------------------
// Fused per-edge-pair pipeline with WMMA. One wave = 32 pairs (two 16-row
// M-tiles sharing every B fragment, so each global_load_b128 feeds 2 WMMAs).
//   ef[64]       = [relu(e*s1+b1) | relu(a*s2+b2)]     (built in A-frag layout)
//   ef2[32][64]  = relu(ef @ edge_emb^T + b)           (WMMA, C->LDS->A relayout)
//   per head h:  f  = leaky(ef2 @ wf_h^T + wfb+bf)     (WMMA)
//                cp = leaky(sum_f af[f]*f + afb)       (packed-f16 epilogue +
//                                                       shfl_xor reduce)
// ---------------------------------------------------------------------------
__global__ __launch_bounds__(256) void edge_cp_kernel(
    const float* __restrict__ edge, const float* __restrict__ adj,
    const float* __restrict__ sctab,
    const _Float16* __restrict__ emb16, const float* __restrict__ embb,
    const _Float16* __restrict__ wf16, const float* __restrict__ fbias,
    const float* __restrict__ afw, const float* __restrict__ afb,
    float* __restrict__ cp, int P)
{
    __shared__ float scs[128];
    __shared__ __align__(16) _Float16 ef2[8][32 * 64];
    const int tid = threadIdx.x;
    if (tid < 128) scs[tid] = sctab[tid];
    __syncthreads();

    const int wave = tid >> 5, lane = tid & 31;
    const int hs = lane >> 4, col = lane & 15;
    const int p0 = (blockIdx.x * 8 + wave) * 32;

    int pairL = p0 + col;      if (pairL >= P) pairL = P - 1;
    int pairH = p0 + 16 + col; if (pairH >= P) pairH = P - 1;
    const float eL = edge[pairL], aL = adj[pairL];
    const float eH = edge[pairH], aH = adj[pairH];

    // A fragments of ef built directly in registers, contiguous k order (same as B):
    // lane holds row=lane&15, k = hs*16 + {0..15}
    v16h aL0, aL1, aH0, aH1;
#pragma unroll
    for (int i2 = 0; i2 < 16; ++i2) {
        const int k = hs * 16 + i2;
        aL0[i2] = (_Float16)fmaxf(fmaf(eL, scs[k], scs[32 + k]), 0.f);
        aL1[i2] = (_Float16)fmaxf(fmaf(aL, scs[64 + k], scs[96 + k]), 0.f);
        aH0[i2] = (_Float16)fmaxf(fmaf(eH, scs[k], scs[32 + k]), 0.f);
        aH1[i2] = (_Float16)fmaxf(fmaf(aH, scs[64 + k], scs[96 + k]), 0.f);
    }

    // edge_emb: 4 N-tiles x 2 K-chunks x 2 M-tiles; relu results to LDS
#pragma unroll
    for (int t = 0; t < 4; ++t) {
        const int f = t * 16 + col;
        const _Float16* wr = emb16 + f * 64 + hs * 16;  // B frag: 16 contiguous halves
        const v16h b0 = *(const v16h*)(wr);
        const v16h b1 = *(const v16h*)(wr + 32);
        const float bb = embb[f];
        v8f c = {};
        c = __builtin_amdgcn_wmma_f32_16x16x32_f16(false, aL0, false, b0, (short)0, c, false, false);
        c = __builtin_amdgcn_wmma_f32_16x16x32_f16(false, aL1, false, b1, (short)0, c, false, false);
#pragma unroll
        for (int r = 0; r < 8; ++r)
            ef2[wave][(r + hs * 8) * 64 + f] = (_Float16)fmaxf(c[r] + bb, 0.f);
        v8f d = {};
        d = __builtin_amdgcn_wmma_f32_16x16x32_f16(false, aH0, false, b0, (short)0, d, false, false);
        d = __builtin_amdgcn_wmma_f32_16x16x32_f16(false, aH1, false, b1, (short)0, d, false, false);
#pragma unroll
        for (int r = 0; r < 8; ++r)
            ef2[wave][(16 + r + hs * 8) * 64 + f] = (_Float16)fmaxf(d[r] + bb, 0.f);
    }
    asm volatile("s_wait_dscnt 0" ::: "memory");

    // reload ef2 as A fragments for both M-tiles (per-wave LDS region)
    v16h eA0, eA1, eB0, eB1;
    {
        const _Float16* base = &ef2[wave][0];
        const int rL = col, rH = 16 + col;
        const v8h l0 = *(const v8h*)(base + rL * 64 + hs * 16);
        const v8h l1 = *(const v8h*)(base + rL * 64 + 8 + hs * 16);
        const v8h l2 = *(const v8h*)(base + rL * 64 + 32 + hs * 16);
        const v8h l3 = *(const v8h*)(base + rL * 64 + 40 + hs * 16);
        const v8h m0 = *(const v8h*)(base + rH * 64 + hs * 16);
        const v8h m1 = *(const v8h*)(base + rH * 64 + 8 + hs * 16);
        const v8h m2 = *(const v8h*)(base + rH * 64 + 32 + hs * 16);
        const v8h m3 = *(const v8h*)(base + rH * 64 + 40 + hs * 16);
#pragma unroll
        for (int i2 = 0; i2 < 8; ++i2) {
            eA0[i2] = l0[i2]; eA0[i2 + 8] = l1[i2];
            eA1[i2] = l2[i2]; eA1[i2 + 8] = l3[i2];
            eB0[i2] = m0[i2]; eB0[i2 + 8] = m1[i2];
            eB1[i2] = m2[i2]; eB1[i2 + 8] = m3[i2];
        }
    }

    const h2 zero2 = {};
    for (int h = 0; h < 12; ++h) {
        if (h < 11) __builtin_prefetch(wf16 + (h + 1) * 4096, 0, 1);
        h2 pL[4], pH[4];
#pragma unroll
        for (int i = 0; i < 4; ++i) { pL[i] = zero2; pH[i] = zero2; }
        const _Float16* wfh = wf16 + h * 4096;
#pragma unroll
        for (int t = 0; t < 4; ++t) {
            const int f = t * 16 + col;
            const _Float16* wr = wfh + f * 64 + hs * 16;
            const v16h b0 = *(const v16h*)(wr);
            const v16h b1 = *(const v16h*)(wr + 32);
            v8f c = {};
            c = __builtin_amdgcn_wmma_f32_16x16x32_f16(false, eA0, false, b0, (short)0, c, false, false);
            c = __builtin_amdgcn_wmma_f32_16x16x32_f16(false, eA1, false, b1, (short)0, c, false, false);
            v8f d = {};
            d = __builtin_amdgcn_wmma_f32_16x16x32_f16(false, eB0, false, b0, (short)0, d, false, false);
            d = __builtin_amdgcn_wmma_f32_16x16x32_f16(false, eB1, false, b1, (short)0, d, false, false);
            // packed-f16 epilogue: leaky(x) = max(x,0) + 0.2*min(x,0),
            // folded into two v_pk_fma_f16 against af and 0.2*af splats
            const _Float16 fbh = (_Float16)fbias[h * 64 + f];
            const float avf = afw[h * 64 + f];
            const h2 fb2 = {fbh, fbh};
            const _Float16 avp1 = (_Float16)avf;
            const _Float16 avn1 = (_Float16)(0.2f * avf);
            const h2 avp = {avp1, avp1};
            const h2 avn = {avn1, avn1};
#pragma unroll
            for (int i = 0; i < 4; ++i) {
                h2 x; x[0] = (_Float16)c[2 * i]; x[1] = (_Float16)c[2 * i + 1];
                x += fb2;
                pL[i] = __builtin_elementwise_fma(avp, __builtin_elementwise_max(x, zero2), pL[i]);
                pL[i] = __builtin_elementwise_fma(avn, __builtin_elementwise_min(x, zero2), pL[i]);
                h2 y; y[0] = (_Float16)d[2 * i]; y[1] = (_Float16)d[2 * i + 1];
                y += fb2;
                pH[i] = __builtin_elementwise_fma(avp, __builtin_elementwise_max(y, zero2), pH[i]);
                pH[i] = __builtin_elementwise_fma(avn, __builtin_elementwise_min(y, zero2), pH[i]);
            }
        }
        // reduce over the 16 columns (low-4-bit xor stays within each half-wave)
#pragma unroll
        for (int m = 1; m < 16; m <<= 1) {
#pragma unroll
            for (int i = 0; i < 4; ++i) {
                pL[i] += __builtin_bit_cast(h2, __shfl_xor(__builtin_bit_cast(float, pL[i]), m, 32));
                pH[i] += __builtin_bit_cast(h2, __shfl_xor(__builtin_bit_cast(float, pH[i]), m, 32));
            }
        }
        if (col == 0) {
            const float ab = afb[h];
            float* cph = cp + (size_t)h * P;
#pragma unroll
            for (int i = 0; i < 4; ++i) {
#pragma unroll
                for (int j = 0; j < 2; ++j) {
                    const int r = 2 * i + j;
                    int pp = p0 + hs * 8 + r;
                    if (pp < P) cph[pp] = lrelu((float)pL[i][j] + ab);
                    pp = p0 + 16 + hs * 8 + r;
                    if (pp < P) cph[pp] = lrelu((float)pH[i][j] + ab);
                }
            }
        }
    }
}

// ---------------------------------------------------------------------------
// node feature pipeline (bandwidth-trivial)
// ---------------------------------------------------------------------------
__global__ void kx_kernel(const float* __restrict__ x_c, const float* __restrict__ x_d,
                          const float* __restrict__ x_tw, const float* __restrict__ x_dual,
                          const float* __restrict__ nr, float* __restrict__ x0, int rows)
{
    const int g = blockIdx.x * blockDim.x + threadIdx.x;
    if (g >= rows * 128) return;
    const int row = g >> 7, k = g & 127;
    float v;
    if (k < 32) {
        v = x_c[row * 2] * nr[k] + x_c[row * 2 + 1] * nr[32 + k] + nr[64 + k];
    } else if (k < 64) {
        const int kk = k - 32;
        v = x_d[row] * nr[96 + kk] + nr[128 + kk];
    } else if (k < 96) {
        const int kk = k - 64;
        v = x_tw[row * 2] * nr[160 + kk] + x_tw[row * 2 + 1] * nr[192 + kk] + nr[224 + kk];
    } else {
        const int kk = k - 96;
        v = x_dual[row] * nr[256 + kk] + nr[288 + kk];
    }
    x0[g] = fmaxf(v, 0.f);
}

__global__ void knf_kernel(const float* __restrict__ x0, const float* __restrict__ w,
                           const float* __restrict__ b, float* __restrict__ nf, int rows)
{
    const int g = blockIdx.x * blockDim.x + threadIdx.x;
    if (g >= rows * 128) return;
    const int row = g >> 7, o = g & 127;
    float acc = b[o];
    const float* xr = x0 + (size_t)row * 128;
    const float* wr = w + (size_t)o * 128;
    for (int k = 0; k < 128; ++k) acc += xr[k] * wr[k];
    nf[g] = fmaxf(acc, 0.f);
}

__global__ void lin_kernel(const float* __restrict__ xin, const float* __restrict__ w,
                           const float* __restrict__ b, float* __restrict__ hv,
                           int ld, int ind)
{
    const int row = blockIdx.x, o = threadIdx.x;
    float acc = b[o];
    const float* xr = xin + (size_t)row * ld;
    const float* wr = w + (size_t)o * ind;
    for (int k = 0; k < ind; ++k) acc += xr[k] * wr[k];
    hv[(size_t)row * 64 + o] = acc;
}

__global__ void a12_kernel(const float* __restrict__ xin, const float* __restrict__ w1r,
                           const float* __restrict__ w2r, const float* __restrict__ c12,
                           float* __restrict__ a1, float* __restrict__ a2, int ld, int ind, int rows)
{
    const int row = blockIdx.x * blockDim.x + threadIdx.x;
    if (row >= rows) return;
    const float* xr = xin + (size_t)row * ld;
    float s1 = c12[0], s2 = c12[1];
    for (int k = 0; k < ind; ++k) { s1 += xr[k] * w1r[k]; s2 += xr[k] * w2r[k]; }
    a1[row] = s1; a2[row] = s2;
}

// ---------------------------------------------------------------------------
// masked softmax over neighbors + aggregation of h
// ---------------------------------------------------------------------------
__global__ __launch_bounds__(256) void softagg_kernel(
    const float* __restrict__ a1, const float* __restrict__ a2,
    const float* __restrict__ ahb, const float* __restrict__ cph,
    const float* __restrict__ adj, const float* __restrict__ hvec,
    float* __restrict__ outx, int ldout, int coloff, int Nn)
{
    __shared__ float sc[512];
    __shared__ float red[256];
    const int row = blockIdx.x;
    const int tid = threadIdx.x;
    const int b = row / Nn;
    const float base = a1[row] + ahb[0];
    float lmax = -3.0e38f;
    for (int m = tid; m < Nn; m += 256) {
        float s = lrelu(base + a2[b * Nn + m]) + cph[(size_t)row * Nn + m];
        if (!(adj[(size_t)row * Nn + m] > 0.f)) s = -9.0e15f;
        sc[m] = s;
        lmax = fmaxf(lmax, s);
    }
    red[tid] = lmax; __syncthreads();
    for (int s = 128; s > 0; s >>= 1) { if (tid < s) red[tid] = fmaxf(red[tid], red[tid + s]); __syncthreads(); }
    const float mx = red[0]; __syncthreads();
    float lsum = 0.f;
    for (int m = tid; m < Nn; m += 256) { const float ev = __expf(sc[m] - mx); sc[m] = ev; lsum += ev; }
    red[tid] = lsum; __syncthreads();
    for (int s = 128; s > 0; s >>= 1) { if (tid < s) red[tid] += red[tid + s]; __syncthreads(); }
    const float inv = 1.f / red[0]; __syncthreads();
    const int o = tid & 63, seg = tid >> 6;
    const int m0 = (Nn * seg) >> 2, m1 = (Nn * (seg + 1)) >> 2;
    float acc = 0.f;
    for (int m = m0; m < m1; ++m) acc += sc[m] * hvec[(size_t)(b * Nn + m) * 64 + o];
    red[tid] = acc; __syncthreads();
    if (tid < 64) {
        const float tot = red[tid] + red[tid + 64] + red[tid + 128] + red[tid + 192];
        outx[(size_t)row * ldout + coloff + tid] = tot * inv;
    }
}

// per-node layernorm over (batch, feature) = 128 values  (assumes B==2)
__global__ void norm_kernel(float* __restrict__ x, int ld, int coloff, int Nn, int lastact)
{
    const int n = blockIdx.x, tid = threadIdx.x;
    const int b = tid >> 6, o = tid & 63;
    const size_t idx = (size_t)(b * Nn + n) * ld + coloff + o;
    const float v = x[idx];
    __shared__ float r1[128], r2[128];
    r1[tid] = v; r2[tid] = v * v; __syncthreads();
    for (int s = 64; s > 0; s >>= 1) { if (tid < s) { r1[tid] += r1[tid + s]; r2[tid] += r2[tid + s]; } __syncthreads(); }
    const float mean = r1[0] * (1.f / 128.f);
    const float var = r2[0] * (1.f / 128.f) - mean * mean;
    float xn = (v - mean) * rsqrtf(1e-5f + var);
    if (lastact) xn = lrelu(xn);
    x[idx] = xn;
}

__global__ void combine_kernel(const float* __restrict__ hp1, float* __restrict__ x2, int rows)
{
    const int g = blockIdx.x * blockDim.x + threadIdx.x;
    if (g >= rows * 64) return;
    const size_t stride = (size_t)rows * 64;
    const float s = 0.25f * (hp1[g] + hp1[stride + g] + hp1[2 * stride + g] + hp1[3 * stride + g]);
    x2[g] = fmaxf(s, 0.f);
}

__global__ void out1_kernel(const float* __restrict__ x2, const float* __restrict__ w,
                            const float* __restrict__ b, float* __restrict__ o1, int rows)
{
    const int g = blockIdx.x * blockDim.x + threadIdx.x;
    if (g >= rows * 32) return;
    const int row = g >> 5, j = g & 31;
    float acc = b[j];
    for (int k = 0; k < 64; ++k) acc += x2[row * 64 + k] * w[j * 64 + k];
    o1[g] = fmaxf(acc, 0.f);
}

__global__ void out2_kernel(const float* __restrict__ o1, const float* __restrict__ w,
                            const float* __restrict__ b, float* __restrict__ out, int rows)
{
    const int row = blockIdx.x * blockDim.x + threadIdx.x;
    if (row >= rows) return;
    float acc = b[0];
    for (int j = 0; j < 32; ++j) acc += o1[row * 32 + j] * w[j];
    out[row] = acc;
}

// ---------------------------------------------------------------------------
extern "C" void kernel_launch(void* const* d_in, const int* in_sizes, int n_in,
                              void* d_out, int out_size, void* d_ws, size_t ws_size,
                              hipStream_t stream)
{
    (void)n_in; (void)out_size; (void)ws_size;
    int ii = 0;
    auto F = [&](void) { return (const float*)d_in[ii++]; };
    const float* x_c = F();  const float* x_d = F();
    const float* x_tw = F(); const float* x_dual = F();
    const float* edge = F(); const float* adj = F();
    const float *nw1w = F(), *nw1b = F(), *nw2w = F(), *nw2b = F();
    const float *nw3w = F(), *nw3b = F(), *nw4w = F(), *nw4b = F();
    const float *ew1w = F(), *ew1b = F(), *ew2w = F(), *ew2b = F();
    const float *nembw = F(), *nembb = F();
    const float *eembw = F(), *eembb = F();
    struct HeadP { const float *whw, *whb, *wh1w, *wh1b, *wh2w, *wh2b, *ahw, *ahb, *wfw, *wfb, *afw, *afb, *bf; };
    HeadP hp[12];
    for (int h = 0; h < 12; ++h) {
        hp[h].whw = F();  hp[h].whb = F();
        hp[h].wh1w = F(); hp[h].wh1b = F();
        hp[h].wh2w = F(); hp[h].wh2b = F();
        hp[h].ahw = F();  hp[h].ahb = F();
        hp[h].wfw = F();  hp[h].wfb = F();
        hp[h].afw = F();  hp[h].afb = F();
        hp[h].bf = F();
    }
    const float *o1w = F(), *o1b = F(), *o2w = F(), *o2b = F();

    const int rows = in_sizes[1];   // B*N  (x_d is [B,N,1])
    const int P = in_sizes[4];      // B*N*N
    const int Nn = P / rows;        // N

    // workspace carve (256B aligned)
    char* wsb = (char*)d_ws;
    size_t cur = 0;
    auto alloc = [&](size_t bytes) -> void* {
        cur = (cur + 255) & ~(size_t)255;
        void* p = wsb + cur;
        cur += bytes;
        return p;
    };
    float*    cp    = (float*)alloc((size_t)12 * P * 4);
    float*    scw   = (float*)alloc(128 * 4);
    float*    nr    = (float*)alloc(320 * 4);
    _Float16* emb16 = (_Float16*)alloc(4096 * 2);
    _Float16* wf16  = (_Float16*)alloc((size_t)12 * 4096 * 2);
    float*    fbias = (float*)alloc(12 * 64 * 4);
    float*    afww  = (float*)alloc(12 * 64 * 4);
    float*    afbw  = (float*)alloc(12 * 4);
    float*    w1r   = (float*)alloc(12 * 512 * 4);
    float*    w2r   = (float*)alloc(12 * 512 * 4);
    float*    c12   = (float*)alloc(12 * 2 * 4);
    float*    x0    = (float*)alloc((size_t)rows * 128 * 4);
    float*    nf    = (float*)alloc((size_t)rows * 128 * 4);
    float*    x1    = (float*)alloc((size_t)rows * 512 * 4);
    float*    hv    = (float*)alloc((size_t)rows * 64 * 4);
    float*    a1w   = (float*)alloc((size_t)rows * 4);
    float*    a2w   = (float*)alloc((size_t)rows * 4);
    float*    hp1   = (float*)alloc((size_t)4 * rows * 64 * 4);
    float*    x2    = (float*)alloc((size_t)rows * 64 * 4);
    float*    o1    = (float*)alloc((size_t)rows * 32 * 4);

    setup_kernel<<<1, 256, 0, stream>>>(ew1w, ew1b, ew2w, ew2b,
                                        nw1w, nw1b, nw2w, nw2b,
                                        nw3w, nw3b, nw4w, nw4b,
                                        eembw, scw, nr, emb16);
    for (int h = 0; h < 12; ++h) {
        const int ind = (h < 8) ? 128 : 512;
        conv_head_kernel<<<1, 256, 0, stream>>>(
            hp[h].wh1w, hp[h].wh1b, hp[h].wh2w, hp[h].wh2b, hp[h].ahw,
            hp[h].wfw, hp[h].wfb, hp[h].bf, hp[h].afw, hp[h].afb, ind,
            wf16 + (size_t)h * 4096, fbias + h * 64, afww + h * 64, afbw + h,
            w1r + h * 512, w2r + h * 512, c12 + h * 2);
    }

    kx_kernel<<<(rows * 128 + 255) / 256, 256, 0, stream>>>(x_c, x_d, x_tw, x_dual, nr, x0, rows);
    knf_kernel<<<(rows * 128 + 255) / 256, 256, 0, stream>>>(x0, nembw, nembb, nf, rows);

    edge_cp_kernel<<<(P + 255) / 256, 256, 0, stream>>>(
        edge, adj, scw, emb16, eembb, wf16, fbias, afww, afbw, cp, P);

    // layer 0: 8 heads, concat + leaky(layernorm)
    for (int h = 0; h < 8; ++h) {
        lin_kernel<<<rows, 64, 0, stream>>>(nf, hp[h].whw, hp[h].whb, hv, 128, 128);
        a12_kernel<<<(rows + 255) / 256, 256, 0, stream>>>(nf, w1r + h * 512, w2r + h * 512,
                                                           c12 + h * 2, a1w, a2w, 128, 128, rows);
        softagg_kernel<<<rows, 256, 0, stream>>>(a1w, a2w, hp[h].ahb, cp + (size_t)h * P,
                                                 adj, hv, x1, 512, h * 64, Nn);
        norm_kernel<<<Nn, 128, 0, stream>>>(x1, 512, h * 64, Nn, 1);
    }

    // layer 1: 4 heads, mean + relu
    for (int h = 0; h < 4; ++h) {
        const int g = 8 + h;
        lin_kernel<<<rows, 64, 0, stream>>>(x1, hp[g].whw, hp[g].whb, hv, 512, 512);
        a12_kernel<<<(rows + 255) / 256, 256, 0, stream>>>(x1, w1r + g * 512, w2r + g * 512,
                                                           c12 + g * 2, a1w, a2w, 512, 512, rows);
        softagg_kernel<<<rows, 256, 0, stream>>>(a1w, a2w, hp[g].ahb, cp + (size_t)g * P,
                                                 adj, hv, hp1 + (size_t)h * rows * 64, 64, 0, Nn);
        norm_kernel<<<Nn, 128, 0, stream>>>(hp1 + (size_t)h * rows * 64, 64, 0, Nn, 0);
    }

    combine_kernel<<<(rows * 64 + 255) / 256, 256, 0, stream>>>(hp1, x2, rows);
    out1_kernel<<<(rows * 32 + 255) / 256, 256, 0, stream>>>(x2, o1w, o1b, o1, rows);
    out2_kernel<<<(rows + 255) / 256, 256, 0, stream>>>(o1, o2w, o2b, (float*)d_out, rows);
}
